// BasicBlock_77884936946099
// MI455X (gfx1250) — matrix operand order repaired and checked by reading, not compiled, not run on hardware
//
#include <hip/hip_runtime.h>
#include <hip/hip_bf16.h>
#include <math.h>

// Problem constants (from reference): B=4, N=4096, K=16 neighbors, C=6
#define NN  4096
#define BBS 4
#define KNB 16
#define CC  6

typedef __attribute__((ext_vector_type(16))) __bf16          v16bf;
typedef __attribute__((ext_vector_type(4)))  __bf16          v4bf;
typedef __attribute__((ext_vector_type(16))) unsigned short  v16us;
typedef __attribute__((ext_vector_type(8)))  float           v8f;
typedef __attribute__((ext_vector_type(4)))  float           v4f;

__device__ __forceinline__ unsigned short f2bf(float f) {
  __bf16 h = (__bf16)f;
  return __builtin_bit_cast(unsigned short, h);
}
__device__ __forceinline__ float bf2f(unsigned short h) {
  return __uint_as_float(((unsigned)h) << 16);
}

// One K-chunk (32 wide) of operands for the WMMA pipeline.
template<int NCT>
struct Chunk {
  v4f   c[4];        // 16 f32 of the A tile this lane owns
  v16us cb[NCT];     // bf16 B fragments (raw bits)
};

template<int NCT>
__device__ __forceinline__ void load_chunk(Chunk<NCT>& ch,
                                           const float* __restrict__ mrow,
                                           const unsigned short* __restrict__ Xt,
                                           int kb, int hi, int rl) {
  // A fragment (16x32 bf16): lane<16 holds K 0-15,
  // lane>=16 holds K 16-31 (matches B's contiguous order).
  const int ka = kb + hi * 16;
  ch.c[0] = *(const v4f*)(mrow + ka);
  ch.c[1] = *(const v4f*)(mrow + ka + 4);
  ch.c[2] = *(const v4f*)(mrow + ka + 8);
  ch.c[3] = *(const v4f*)(mrow + ka + 12);
  // B fragment (32x16 bf16): lane = column, 16 consecutive K = 32 bytes.
  const int koff = kb + hi * 16;
#pragma unroll
  for (int t = 0; t < NCT; ++t)
    ch.cb[t] = *(const v16us*)(Xt + (size_t)(t * 16 + rl) * NN + koff);
}

template<int NCT>
__device__ __forceinline__ void mma_chunk(const Chunk<NCT>& ch, v8f (&acc)[NCT]) {
  union { v16bf v; v4bf q[4]; } ua;
#pragma unroll
  for (int i = 0; i < 4; ++i)
    ua.q[i] = __builtin_convertvector(ch.c[i], v4bf);   // paired v_cvt_pk_bf16_f32
  const v16bf a = ua.v;
#pragma unroll
  for (int t = 0; t < NCT; ++t)
    acc[t] = __builtin_amdgcn_wmma_f32_16x16x32_bf16(
        false, a, false, __builtin_bit_cast(v16bf, ch.cb[t]),
        (short)0, acc[t], false, false);
}

// ---------------------------------------------------------------------------
// Skinny GEMM: Y[n, j] = sum_m M[n,m] * Xt[j, m]  (M: NxN f32, Xt: bf16,
// j-major, row stride NN). 4 waves per block, each wave owns a K quarter,
// 2x-unrolled ping-pong software pipeline under v_wmma_f32_16x16x32_bf16,
// partials reduced through LDS, fused epilogue:
//   MODE 0: raw f32 out[j][n] for j<4 (matvec A.x / L.x)
//   MODE 1: out = relu(Y @ W + bias) -> bf16 Xt_out[b*cout+co][n]
//   MODE 2: same math, f32 write into scales[b][n][sidx][co]
// ---------------------------------------------------------------------------
template<int NCT, int MODE>
__global__ __launch_bounds__(128)
void gemm_bf16_wmma(const float* __restrict__ M,
                    const unsigned short* __restrict__ Xt,
                    int cin, int cout,
                    const float* __restrict__ W,
                    const float* __restrict__ bias,
                    unsigned short* __restrict__ xt_out,
                    float* __restrict__ f_out,
                    int sidx) {
  __shared__ float lds[4 * 16 * 64];
  const int tid  = threadIdx.x;
  const int wv   = tid >> 5;           // wave id: K-quarter selector
  const int lane = tid & 31;
  const int hi   = lane >> 4;
  const int rl   = lane & 15;
  const float* mrow = M + (size_t)(blockIdx.x * 16 + rl) * NN;

  constexpr int KQ = NN / 4;           // 1024 = 32 chunks of 32
  const int k0 = wv * KQ;

  v8f acc[NCT] = {};
  Chunk<NCT> s0, s1;

  load_chunk<NCT>(s0, mrow, Xt, k0, hi, rl);          // chunk 0

#pragma unroll 1
  for (int it = 0; it < KQ / 64 - 1; ++it) {          // 15 steady iterations
    const int base = k0 + it * 64;
    load_chunk<NCT>(s1, mrow, Xt, base + 32, hi, rl); // chunk 2it+1
    mma_chunk<NCT>(s0, acc);                          // chunk 2it
    load_chunk<NCT>(s0, mrow, Xt, base + 64, hi, rl); // chunk 2it+2
    mma_chunk<NCT>(s1, acc);                          // chunk 2it+1
  }
  // tail: chunks 30 (already in s0) and 31
  load_chunk<NCT>(s1, mrow, Xt, k0 + KQ - 32, hi, rl);
  mma_chunk<NCT>(s0, acc);
  mma_chunk<NCT>(s1, acc);

  // C layout: VGPR i <-> row (i + hi*8), lane&15 <-> column. Per-wave spill.
  float* myl = lds + wv * (16 * 64);
#pragma unroll
  for (int t = 0; t < NCT; ++t)
#pragma unroll
    for (int i = 0; i < 8; ++i)
      myl[(i + hi * 8) * 64 + t * 16 + rl] = acc[t][i];
  __syncthreads();

  // Fused epilogue: 64 (row, batch) pairs; sum the 4 K-quarter partials.
  if (tid < 64) {
    const int r  = tid >> 2;
    const int bb = tid & 3;
    const int n  = blockIdx.x * 16 + r;
    if (MODE == 0) {
      const int o = r * 64 + bb;
      f_out[(size_t)bb * NN + n] =
          lds[o] + lds[1024 + o] + lds[2048 + o] + lds[3072 + o];
    } else {
      float yv[16];
      for (int ci = 0; ci < cin; ++ci) {
        const int o = r * 64 + bb * cin + ci;
        yv[ci] = lds[o] + lds[1024 + o] + lds[2048 + o] + lds[3072 + o];
      }
      for (int co = 0; co < cout; ++co) {
        float v = bias[co];
        for (int ci = 0; ci < cin; ++ci) v += yv[ci] * W[ci * cout + co];
        v = fmaxf(v, 0.0f);
        if (MODE == 1) {
          xt_out[(size_t)(bb * cout + co) * NN + n] = f2bf(v);
        } else {
          f_out[(((size_t)bb * NN + n) * 4 + sidx) * CC + co] = v;
        }
      }
    }
  }
}

// ---------------------------------------------------------------------------
// Elementwise helper kernels
// ---------------------------------------------------------------------------
__global__ void k_build_x(const float* __restrict__ x,
                          unsigned short* __restrict__ xt) {
  int i = blockIdx.x * blockDim.x + threadIdx.x;
  if (i < BBS * NN) xt[i] = f2bf(x[i]);   // Xt_x[j=b][m=n]
}

__global__ void k_build_feat0(const float* __restrict__ x,
                              const float* __restrict__ bv,
                              const float* __restrict__ rawA,
                              const float* __restrict__ rawL,
                              unsigned short* __restrict__ xt) {
  int i = blockIdx.x * blockDim.x + threadIdx.x;
  if (i >= BBS * NN) return;
  int b = i / NN, n = i % NN;
  xt[(size_t)(b * 3 + 0) * NN + n] = f2bf(x[i]);
  xt[(size_t)(b * 3 + 1) * NN + n] = f2bf(rawA[i] - bv[i]);
  xt[(size_t)(b * 3 + 2) * NN + n] = f2bf(rawL[i]);
}

// kNN-attention + gradient MLP + softplus threshold + soft shrinkage
__global__ void k_final(const float* __restrict__ x,
                        const int* __restrict__ knn,
                        const float* __restrict__ sens,
                        const float* __restrict__ scales,
                        const unsigned short* __restrict__ xt_feat,
                        const float* __restrict__ W1, const float* __restrict__ b1,
                        const float* __restrict__ W2, const float* __restrict__ b2,
                        const float* __restrict__ Wt, const float* __restrict__ bt,
                        const float* __restrict__ alpha_p,
                        float* __restrict__ out) {
  int i = blockIdx.x * blockDim.x + threadIdx.x;
  if (i >= BBS * NN) return;
  int b = i / NN, n = i % NN;

  const float* sb = scales + ((size_t)b * NN + n) * 4 * CC;
  float q[CC];
#pragma unroll
  for (int c = 0; c < CC; ++c)
    q[c] = 0.25f * (sb[c] + sb[CC + c] + sb[2 * CC + c] + sb[3 * CC + c]);

  float agg[4][CC] = {};
  for (int k = 0; k < KNB; ++k) {
    int m = knn[n * KNB + k];
    const float* nb = scales + ((size_t)b * NN + m) * 4 * CC;
#pragma unroll
    for (int s = 0; s < 4; ++s)
#pragma unroll
      for (int c = 0; c < CC; ++c)
        agg[s][c] += nb[s * CC + c];
  }

  const float inv = 1.0f / (float)KNB;
  float sc[4], mx = -1e30f;
#pragma unroll
  for (int s = 0; s < 4; ++s) {
    float d = 0.0f;
#pragma unroll
    for (int c = 0; c < CC; ++c) { agg[s][c] *= inv; d += q[c] * agg[s][c]; }
    sc[s] = d * 0.4082482904638631f;      // 1/sqrt(C)
    mx = fmaxf(mx, sc[s]);
  }
  float wgt[4], sum = 0.0f;
#pragma unroll
  for (int s = 0; s < 4; ++s) { wgt[s] = __expf(sc[s] - mx); sum += wgt[s]; }
  const float rs = 1.0f / sum;
  const float sw = sens[n];

  float fa[CC];
#pragma unroll
  for (int c = 0; c < CC; ++c) {
    float v = 0.0f;
#pragma unroll
    for (int s = 0; s < 4; ++s) v += wgt[s] * agg[s][c];
    fa[c] = v * rs * sw;
  }

  float h1[3];
#pragma unroll
  for (int j = 0; j < 3; ++j) {
    float v = b1[j];
#pragma unroll
    for (int c = 0; c < CC; ++c) v += fa[c] * W1[c * 3 + j];
    h1[j] = fmaxf(v, 0.0f);
  }
  float g = b2[0];
#pragma unroll
  for (int j = 0; j < 3; ++j) g += h1[j] * W2[j];

  float tv = bt[0];
#pragma unroll
  for (int c = 0; c < CC; ++c)
    tv += bf2f(xt_feat[(size_t)(b * CC + c) * NN + n]) * Wt[c];
  float thr = (tv > 20.0f) ? tv : log1pf(__expf(tv));

  float z = x[i] - alpha_p[0] * g;
  float az = fabsf(z) - thr;
  out[i] = (az > 0.0f) ? copysignf(az, z) : 0.0f;
}

// ---------------------------------------------------------------------------
extern "C" void kernel_launch(void* const* d_in, const int* in_sizes, int n_in,
                              void* d_out, int out_size, void* d_ws, size_t ws_size,
                              hipStream_t stream) {
  (void)in_sizes; (void)n_in; (void)out_size; (void)ws_size;
  const float* x    = (const float*)d_in[0];
  const float* bv   = (const float*)d_in[1];
  const float* L    = (const float*)d_in[2];
  const float* A    = (const float*)d_in[3];
  const float* Li[4] = {(const float*)d_in[4], (const float*)d_in[5],
                        (const float*)d_in[6], (const float*)d_in[7]};
  const int*   knn  = (const int*)d_in[8];
  const float* sens = (const float*)d_in[9];
  const float* gW[4] = {(const float*)d_in[10], (const float*)d_in[12],
                        (const float*)d_in[14], (const float*)d_in[16]};
  const float* gB[4] = {(const float*)d_in[11], (const float*)d_in[13],
                        (const float*)d_in[15], (const float*)d_in[17]};
  // ms branches at d_in[18 + 4*s + {0..3}], gp1/gp2/thr/alpha at 34..40.

  char* w = (char*)d_ws;
  unsigned short* Xt0 = (unsigned short*)(w);
  unsigned short* Xt1 = (unsigned short*)(w + (size_t)512 * 1024);
  unsigned short* Xt2 = (unsigned short*)(w + (size_t)1024 * 1024);
  unsigned short* Xt3 = (unsigned short*)(w + (size_t)1536 * 1024);
  float* scales = (float*)(w + (size_t)2048 * 1024);          // 1.5 MB
  float* rawA   = (float*)(w + (size_t)2048 * 1024);          // overlaps scales
  float* rawL   = rawA + (size_t)BBS * NN;                    // (dead before scales)

  const dim3 gE((BBS * NN + 255) / 256), bE(256);
  const dim3 gG(NN / 16), bG(128);

  // feat0 = [x, A.x - b, L.x]  (matvecs via WMMA, 4 columns)
  k_build_x<<<gE, bE, 0, stream>>>(x, Xt0);
  gemm_bf16_wmma<1, 0><<<gG, bG, 0, stream>>>(A, Xt0, 0, 0, nullptr, nullptr, nullptr, rawA, 0);
  gemm_bf16_wmma<1, 0><<<gG, bG, 0, stream>>>(L, Xt0, 0, 0, nullptr, nullptr, nullptr, rawL, 0);
  k_build_feat0<<<gE, bE, 0, stream>>>(x, bv, rawA, rawL, Xt1);

  // GCN sequence: 3 -> 8 -> 16 -> 8 -> 6 channels, all against L
  gemm_bf16_wmma<1, 1><<<gG, bG, 0, stream>>>(L, Xt1, 3,  8,  gW[0], gB[0], Xt2, nullptr, 0);
  gemm_bf16_wmma<2, 1><<<gG, bG, 0, stream>>>(L, Xt2, 8,  16, gW[1], gB[1], Xt3, nullptr, 0);
  gemm_bf16_wmma<4, 1><<<gG, bG, 0, stream>>>(L, Xt3, 16, 8,  gW[2], gB[2], Xt0, nullptr, 0);
  gemm_bf16_wmma<2, 1><<<gG, bG, 0, stream>>>(L, Xt0, 8,  6,  gW[3], gB[3], Xt1, nullptr, 0);

  // Multiscale branches: two 6->6 GCN layers each against Li; writes scales
  for (int s = 0; s < 4; ++s) {
    const float* W0 = (const float*)d_in[18 + 4 * s + 0];
    const float* B0 = (const float*)d_in[18 + 4 * s + 1];
    const float* W1 = (const float*)d_in[18 + 4 * s + 2];
    const float* B1 = (const float*)d_in[18 + 4 * s + 3];
    gemm_bf16_wmma<2, 1><<<gG, bG, 0, stream>>>(Li[s], Xt1, 6, 6, W0, B0, Xt2, nullptr, 0);
    gemm_bf16_wmma<2, 2><<<gG, bG, 0, stream>>>(Li[s], Xt2, 6, 6, W1, B1, nullptr, scales, s);
  }

  // Attention over kNN + gradient MLP + threshold + shrinkage
  k_final<<<gE, bE, 0, stream>>>(x, knn, sens, scales, Xt1,
                                 (const float*)d_in[34], (const float*)d_in[35],
                                 (const float*)d_in[36], (const float*)d_in[37],
                                 (const float*)d_in[38], (const float*)d_in[39],
                                 (const float*)d_in[40], (float*)d_out);
}
